// DTransformer_38654705664446
// MI455X (gfx1250) — matrix operand order repaired and checked by reading, not compiled, not run on hardware
//
#include <hip/hip_runtime.h>
#include <hip/hip_bf16.h>
#include <stddef.h>

// ---------------------------------------------------------------------------
// DTransformer forward on gfx1250 (CDNA5).
// All matmuls use v_wmma_f32_16x16x32_f16 (f16 inputs, f32 accumulate).
// Attention is fused per (batch, head): the full 256x256 f32 score matrix
// lives in LDS (256KB <= 320KB WGP LDS), so QK^T -> softmax -> distance
// decay -> softmax -> maxout -> P@V never touches HBM for scores except
// the required q_scores / k_scores outputs.
// ---------------------------------------------------------------------------

typedef _Float16 v16h __attribute__((ext_vector_type(16)));
typedef _Float16 v8h  __attribute__((ext_vector_type(8)));
typedef float    v8f  __attribute__((ext_vector_type(8)));

#define BSd   4
#define SEQd  256
#define Dd    256
#define Hd    8
#define DKd   32
#define NKd   16

__device__ __forceinline__ v16h cat8(v8h lo, v8h hi) {
  return __builtin_shufflevector(lo, hi, 0,1,2,3,4,5,6,7,8,9,10,11,12,13,14,15);
}

__device__ __forceinline__ float wredmax(float v) {
  #pragma unroll
  for (int d = 16; d > 0; d >>= 1) v = fmaxf(v, __shfl_xor(v, d, 32));
  return v;
}
__device__ __forceinline__ float wredsum(float v) {
  #pragma unroll
  for (int d = 16; d > 0; d >>= 1) v += __shfl_xor(v, d, 32);
  return v;
}
__device__ __forceinline__ float wincscan(float v) {  // inclusive prefix over wave32
  int lane = threadIdx.x & 31;
  #pragma unroll
  for (int d = 1; d < 32; d <<= 1) {
    float o = __shfl_up(v, d, 32);
    if (lane >= d) v += o;
  }
  return v;
}

// -------------------------------- casts ------------------------------------
__global__ void k_cast(const float* __restrict__ in, _Float16* __restrict__ out, int n) {
  int i = blockIdx.x * 256 + threadIdx.x;
  if (i < n) out[i] = (_Float16)in[i];
}

// W [256,256] row-major -> Wt (f16) [N=256, K=256]  (Wt[n][k] = W[k][n])
__global__ void k_wt(const float* __restrict__ W, _Float16* __restrict__ Wt) {
  int i = blockIdx.x * 256 + threadIdx.x;   // 65536 threads
  int n = i >> 8, k = i & 255;
  Wt[(size_t)n * 256 + k] = (_Float16)W[(size_t)k * 256 + n];
}

// -------------------------------- GEMM -------------------------------------
// C[M,256] = A[M,256](f16) @ W + bias ; W given transposed (Bt = W^T, [256,256] f16)
// One wave per 16x16 output tile, K=256 -> 8 WMMA steps.
__global__ __launch_bounds__(256) void k_gemm(
    const _Float16* __restrict__ A, const _Float16* __restrict__ Bt,
    const float* __restrict__ bias,
    float* __restrict__ outF, _Float16* __restrict__ outH, int M) {
  const int K = 256, N = 256;
  int wave  = blockIdx.x * 8 + (threadIdx.x >> 5);
  int tiles = (M >> 4) * (N >> 4);
  if (wave >= tiles) return;
  int mT = wave / (N >> 4), nT = wave % (N >> 4);
  int lane = threadIdx.x & 31;
  int hl   = lane & 15;
  int hi16 = lane >> 4;

  const _Float16* Arow = A  + (size_t)(mT * 16 + hl) * K;
  const _Float16* Brow = Bt + (size_t)(nT * 16 + hl) * K;

  v8f acc = {};
  #pragma unroll
  for (int kk = 0; kk < K; kk += 32) {
    int a0 = kk + (hi16 ? 8 : 0);
    v8h alo = *(const v8h*)(Arow + a0);
    v8h ahi = *(const v8h*)(Arow + a0 + 16);
    int b0 = kk + (hi16 ? 8 : 0);
    v8h blo = *(const v8h*)(Brow + b0);
    v8h bhi = *(const v8h*)(Brow + b0 + 16);
    acc = __builtin_amdgcn_wmma_f32_16x16x32_f16(
        false, cat8(alo, ahi), false, cat8(blo, bhi), (short)0, acc, false, false);
  }
  int col = nT * 16 + hl;
  float bv = bias ? bias[col] : 0.0f;
  #pragma unroll
  for (int r = 0; r < 8; ++r) {
    int row = mT * 16 + r + (hi16 ? 8 : 0);
    float v = acc[r] + bv;
    if (outF) outF[(size_t)row * N + col] = v;
    if (outH) outH[(size_t)row * N + col] = (_Float16)v;
  }
}

// ------------------------------ attention ----------------------------------
// One workgroup (8 waves) per (batch, head). Dynamic LDS: 256*256 f32.
// Q: qKnow ? [NK,256] (row = bt%NK, independent of seq pos) : [Bt*256,256]
// K,V: [(Bt/kvDiv)*256, 256]; head h occupies columns h*32..h*32+31.
// scoresMode: 0=none, 1=q_scores [4,8,256,256], 2=k_scores [4,8,256,16,256]
__global__ __launch_bounds__(256) void k_attn(
    const _Float16* __restrict__ Q, const _Float16* __restrict__ Km,
    const _Float16* __restrict__ Vm, const float* __restrict__ gammas,
    float* __restrict__ Out, float* __restrict__ scoresOut,
    int Bt, int kvDiv, int qKnow, int peek, int maxout, int scoresMode) {
  extern __shared__ float S[];   // [256][256]
  int bh = blockIdx.x;
  int bt = bh / Hd, hh = bh % Hd;
  int kb = bt / kvDiv;
  int qn = bt % NKd;
  int tid = threadIdx.x, wv = tid >> 5, lane = tid & 31;
  int hl = lane & 15, hi16 = lane >> 4;
  const float scale = 0.1767766952966369f;   // 1/sqrt(32)

  // ---- Phase 1: S = QK^T * scale (dk=32 -> one WMMA per 16x16 tile) ----
  const _Float16* qbaseKnow = Q + (size_t)qn * Dd + hh * DKd;
  for (int t = wv; t < 256; t += 8) {
    int qT = t >> 4, kT = t & 15;
    int qrow = qT * 16 + hl;
    const _Float16* qp = qKnow ? qbaseKnow
                               : Q + ((size_t)bt * SEQd + qrow) * Dd + hh * DKd;
    int a0 = hi16 ? 8 : 0;
    v8h alo = *(const v8h*)(qp + a0);
    v8h ahi = *(const v8h*)(qp + a0 + 16);
    int key = kT * 16 + hl;
    const _Float16* kp = Km + ((size_t)kb * SEQd + key) * Dd + hh * DKd;
    int b0 = hi16 ? 8 : 0;
    v8h blo = *(const v8h*)(kp + b0);
    v8h bhi = *(const v8h*)(kp + b0 + 16);
    v8f acc = {};
    acc = __builtin_amdgcn_wmma_f32_16x16x32_f16(
        false, cat8(alo, ahi), false, cat8(blo, bhi), (short)0, acc, false, false);
    #pragma unroll
    for (int r = 0; r < 8; ++r) {
      int row = qT * 16 + r + (hi16 ? 8 : 0);
      S[row * 256 + kT * 16 + hl] = acc[r] * scale;
    }
  }
  __syncthreads();

  // ---- Phase 2: per-row softmax -> distance decay -> softmax -> maxout ----
  float g = -fabsf(gammas[hh]);
  for (int q = wv; q < 256; q += 8) {       // one wave per row
    float* row = S + q * 256;
    int c0 = lane * 8;
    float s0[8]; bool mk[8];
    #pragma unroll
    for (int j = 0; j < 8; ++j) {
      s0[j] = row[c0 + j];
      int c = c0 + j;
      mk[j] = peek ? (c <= q) : (c < q);
    }
    // masked softmax #1 -> s_ (defined for all cols; masked ~ 0)
    float mv = -3.4e38f, mval[8];
    #pragma unroll
    for (int j = 0; j < 8; ++j) { mval[j] = mk[j] ? s0[j] : -1e32f; mv = fmaxf(mv, mval[j]); }
    mv = wredmax(mv);
    float e[8], ssum = 0.0f;
    #pragma unroll
    for (int j = 0; j < 8; ++j) { e[j] = expf(mval[j] - mv); ssum += e[j]; }
    ssum = wredsum(ssum);
    float inv = 1.0f / ssum;
    // inclusive cumsum of s_ across the row (lane-local prefix + wave scan)
    float cum[8], run = 0.0f;
    #pragma unroll
    for (int j = 0; j < 8; ++j) { run += e[j] * inv; cum[j] = run; }
    float excl = wincscan(run) - run;
    // decay: dist = sqrt(clip((total-cum)*|q-k|)); total == 1
    float s2[8], mv2 = -3.4e38f;
    #pragma unroll
    for (int j = 0; j < 8; ++j) {
      float pos  = fabsf((float)(q - (c0 + j)));
      float dist = sqrtf(fmaxf((1.0f - (cum[j] + excl)) * pos, 0.0f));
      float eff  = fminf(fmaxf(expf(dist * g), 1e-5f), 1e5f);
      float v    = s0[j] * eff;
      s2[j] = mk[j] ? v : -1e32f;
      mv2 = fmaxf(mv2, s2[j]);
    }
    mv2 = wredmax(mv2);
    float p[8], sum2 = 0.0f;
    #pragma unroll
    for (int j = 0; j < 8; ++j) { p[j] = expf(s2[j] - mv2); sum2 += p[j]; }
    sum2 = wredsum(sum2);
    float inv2 = 1.0f / sum2;
    #pragma unroll
    for (int j = 0; j < 8; ++j) p[j] = mk[j] ? p[j] * inv2 : 0.0f;
    if (maxout) {
      float rm = 0.0f;
      #pragma unroll
      for (int j = 0; j < 8; ++j) rm = fmaxf(rm, p[j]);
      rm = wredmax(rm);
      float sc = fminf(1.0f / (rm + 1e-8f), 5.0f);
      #pragma unroll
      for (int j = 0; j < 8; ++j) p[j] *= sc;
    }
    #pragma unroll
    for (int j = 0; j < 8; ++j) row[c0 + j] = p[j];
    if (scoresMode == 1) {
      float* sp = scoresOut + (((size_t)(bt * Hd + hh) * SEQd + q) * SEQd + c0);
      #pragma unroll
      for (int j = 0; j < 8; ++j) sp[j] = p[j];
    } else if (scoresMode == 2) {
      int b4 = bt >> 4, nn = bt & 15;
      float* sp = scoresOut +
          ((((size_t)(b4 * Hd + hh) * SEQd + q) * NKd + nn) * SEQd + c0);
      #pragma unroll
      for (int j = 0; j < 8; ++j) sp[j] = p[j];
    }
  }
  __syncthreads();

  // ---- Phase 3: O = P @ V (P in LDS f32 -> f16 frags) ----
  for (int t = wv; t < 32; t += 8) {
    int qT = t >> 1, dT = t & 1;
    int qrow = qT * 16 + hl;
    int col  = dT * 16 + hl;     // within head
    v8f acc = {};
    #pragma unroll
    for (int kk = 0; kk < 256; kk += 32) {
      int k0 = kk + (hi16 ? 8 : 0);
      v16h af;
      #pragma unroll
      for (int j = 0; j < 8; ++j) {
        af[j]     = (_Float16)S[qrow * 256 + k0 + j];
        af[j + 8] = (_Float16)S[qrow * 256 + k0 + 16 + j];
      }
      int kb0 = kk + (hi16 ? 8 : 0);
      const _Float16* vp = Vm + (size_t)kb * SEQd * Dd + hh * DKd + col;
      v16h bf;
      #pragma unroll
      for (int j = 0; j < 16; ++j) bf[j] = vp[(size_t)(kb0 + j + (j >= 8 ? 8 : 0)) * Dd];
      acc = __builtin_amdgcn_wmma_f32_16x16x32_f16(
          false, af, false, bf, (short)0, acc, false, false);
    }
    #pragma unroll
    for (int r = 0; r < 8; ++r) {
      int row = qT * 16 + r + (hi16 ? 8 : 0);
      Out[((size_t)bt * SEQd + row) * Dd + hh * DKd + col] = acc[r];
    }
  }
}

// --------------------------- residual + LayerNorm ---------------------------
// mode 0: out[row] = LN(X[row] + O[row]); optional f16 copy.
// mode 1: X = know_params[16,256]; rows index (b*16+n)*256+s; output scattered
//         into z [4,256,4096] at [b,s,n*256+c].
__global__ __launch_bounds__(256) void k_ln(
    const float* __restrict__ X, const float* __restrict__ O,
    const float* __restrict__ g, const float* __restrict__ b,
    float* __restrict__ outF, _Float16* __restrict__ outH, int Rows, int mode) {
  int row = blockIdx.x * 8 + (threadIdx.x >> 5);
  if (row >= Rows) return;
  int lane = threadIdx.x & 31;
  int c0 = lane * 8;
  const float* xr = (mode == 0) ? X + (size_t)row * Dd
                                : X + (size_t)((row >> 8) & 15) * Dd;
  const float* orow = O + (size_t)row * Dd;
  float x[8], s = 0.0f, s2 = 0.0f;
  #pragma unroll
  for (int j = 0; j < 8; ++j) {
    x[j] = xr[c0 + j] + orow[c0 + j];
    s += x[j]; s2 += x[j] * x[j];
  }
  s = wredsum(s); s2 = wredsum(s2);
  float mean = s * (1.0f / Dd);
  float var  = s2 * (1.0f / Dd) - mean * mean;
  float invs = rsqrtf(var + 1e-5f);
  float* op;
  if (mode == 0) op = outF + (size_t)row * Dd;
  else {
    int b4 = row >> 12, nn = (row >> 8) & 15, sq = row & 255;
    op = outF + ((size_t)(b4 * SEQd + sq)) * (NKd * Dd) + nn * Dd;
  }
  #pragma unroll
  for (int j = 0; j < 8; ++j) {
    float y = (x[j] - mean) * invs * g[c0 + j] + b[c0 + j];
    op[c0 + j] = y;
    if (outH) outH[(size_t)row * Dd + c0 + j] = (_Float16)y;
  }
}

// ---------------------------------------------------------------------------
extern "C" void kernel_launch(void* const* d_in, const int* in_sizes, int n_in,
                              void* d_out, int out_size, void* d_ws, size_t ws_size,
                              hipStream_t stream) {
  (void)in_sizes; (void)n_in; (void)out_size;
  const float* q_emb = (const float*)d_in[0];
  const float* s_emb = (const float*)d_in[1];
  const float* know  = (const float*)d_in[3];
  // per-block params: [Wq,bq,(Wk,bk for blk4),Wv,bv,Wo,bo,gam,lng,lnb]
  const float* bq1 = (const float*)d_in[5];  const float* bv1 = (const float*)d_in[7];
  const float* bo1 = (const float*)d_in[9];  const float* gm1 = (const float*)d_in[10];
  const float* lg1 = (const float*)d_in[11]; const float* lb1 = (const float*)d_in[12];
  const float* bq2 = (const float*)d_in[14]; const float* bv2 = (const float*)d_in[16];
  const float* bo2 = (const float*)d_in[18]; const float* gm2 = (const float*)d_in[19];
  const float* lg2 = (const float*)d_in[20]; const float* lb2 = (const float*)d_in[21];
  const float* bq3 = (const float*)d_in[23]; const float* bv3 = (const float*)d_in[25];
  const float* bo3 = (const float*)d_in[27]; const float* gm3 = (const float*)d_in[28];
  const float* lg3 = (const float*)d_in[29]; const float* lb3 = (const float*)d_in[30];
  const float* bq4 = (const float*)d_in[32]; const float* bk4 = (const float*)d_in[34];
  const float* bv4 = (const float*)d_in[36]; const float* bo4 = (const float*)d_in[38];
  const float* gm4 = (const float*)d_in[39]; const float* lg4 = (const float*)d_in[40];
  const float* lb4 = (const float*)d_in[41];

  float* out = (float*)d_out;
  const size_t Z_LEN  = (size_t)BSd * SEQd * NKd * Dd;        // 4,194,304
  const size_t QS_LEN = (size_t)BSd * Hd * SEQd * SEQd;       // 2,097,152
  float* z_out  = out;
  float* qs_out = out + Z_LEN;
  float* ks_out = out + Z_LEN + QS_LEN;

  // bump allocator over d_ws
  char* wsp = (char*)d_ws;
  auto alloc = [&](size_t bytes) -> void* {
    void* p = (void*)wsp;
    wsp += (bytes + 255) & ~(size_t)255;
    return p;
  };

  // f16 transposed weights
  const int WIDX[13] = {4, 6, 8, 13, 15, 17, 22, 24, 26, 31, 33, 35, 37};
  _Float16* wt[13];
  for (int i = 0; i < 13; ++i) {
    wt[i] = (_Float16*)alloc((size_t)Dd * Dd * 2);
    k_wt<<<256, 256, 0, stream>>>((const float*)d_in[WIDX[i]], wt[i]);
  }
  _Float16 *Wq1t = wt[0], *Wv1t = wt[1], *Wo1t = wt[2];
  _Float16 *Wq2t = wt[3], *Wv2t = wt[4], *Wo2t = wt[5];
  _Float16 *Wq3t = wt[6], *Wv3t = wt[7], *Wo3t = wt[8];
  _Float16 *Wq4t = wt[9], *Wk4t = wt[10], *Wv4t = wt[11], *Wo4t = wt[12];

  const int M13 = BSd * SEQd;                 // 1024
  const size_t ACT  = (size_t)M13 * Dd;       // 262144 elems
  _Float16* x16a   = (_Float16*)alloc(ACT * 2);
  _Float16* q16    = (_Float16*)alloc(ACT * 2);
  _Float16* v16    = (_Float16*)alloc(ACT * 2);
  float*    attnF  = (float*)alloc(ACT * 4);
  _Float16* attn16 = (_Float16*)alloc(ACT * 2);
  float*    oF     = (float*)alloc(ACT * 4);
  float*    hq     = (float*)alloc(ACT * 4);
  float*    hs     = (float*)alloc(ACT * 4);
  float*    pblk   = (float*)alloc(ACT * 4);
  _Float16* hq16   = (_Float16*)alloc(ACT * 2);
  _Float16* hs16   = (_Float16*)alloc(ACT * 2);
  _Float16* p16    = (_Float16*)alloc(ACT * 2);
  _Float16* know16 = (_Float16*)alloc((size_t)NKd * Dd * 2);
  _Float16* qkn16  = (_Float16*)alloc((size_t)NKd * Dd * 2);
  _Float16* k416   = (_Float16*)alloc(ACT * 2);
  _Float16* v416   = (_Float16*)alloc(ACT * 2);
  const int M4 = BSd * NKd * SEQd;            // 16384
  const size_t ACT4 = (size_t)M4 * Dd;        // 4,194,304 elems
  float*    big    = (float*)alloc(ACT4 * 4); // attn4F, later reused as o4F
  _Float16* attn416 = (_Float16*)alloc(ACT4 * 2);

  const size_t SMEM = (size_t)SEQd * SEQd * 4;  // 256 KB scores in LDS

  // self-attention block for blocks 1-3 (xk == xq, kq_same)
  auto runBlock = [&](const float* xq, const _Float16* xq16f, const _Float16* xv16f,
                      const float* bq, _Float16* Wqt, const float* bv, _Float16* Wvt,
                      const float* bo, _Float16* Wot, const float* gam,
                      const float* lg, const float* lb,
                      float* scoresPtr, int scoresMode,
                      float* hOut, _Float16* hOut16) {
    k_gemm<<<128, 256, 0, stream>>>(xq16f, Wqt, bq, nullptr, q16, M13);
    k_gemm<<<128, 256, 0, stream>>>(xv16f, Wvt, bv, nullptr, v16, M13);
    k_attn<<<BSd * Hd, 256, SMEM, stream>>>(q16, q16, v16, gam, attnF, scoresPtr,
                                            BSd, 1, 0, 1, 0, scoresMode);
    k_cast<<<(int)((ACT + 255) / 256), 256, 0, stream>>>(attnF, attn16, (int)ACT);
    k_gemm<<<128, 256, 0, stream>>>(attn16, Wot, bo, oF, nullptr, M13);
    k_ln<<<128, 256, 0, stream>>>(xq, oF, lg, lb, hOut, hOut16, M13, 0);
  };

  // Block 1: hq = blk(q_emb)
  k_cast<<<(int)((ACT + 255) / 256), 256, 0, stream>>>(q_emb, x16a, (int)ACT);
  runBlock(q_emb, x16a, x16a, bq1, Wq1t, bv1, Wv1t, bo1, Wo1t, gm1, lg1, lb1,
           nullptr, 0, hq, hq16);
  // Block 2: hs = blk(s_emb)
  k_cast<<<(int)((ACT + 255) / 256), 256, 0, stream>>>(s_emb, x16a, (int)ACT);
  runBlock(s_emb, x16a, x16a, bq2, Wq2t, bv2, Wv2t, bo2, Wo2t, gm2, lg2, lb2,
           nullptr, 0, hs, hs16);
  // Block 3: p = blk(hq, hq, hs)  + q_scores output
  runBlock(hq, hq16, hs16, bq3, Wq3t, bv3, Wv3t, bo3, Wo3t, gm3, lg3, lb3,
           qs_out, 1, pblk, p16);

  // Block 4: knowledge queries (broadcast), K from hq, V from p
  k_cast<<<16, 256, 0, stream>>>(know, know16, NKd * Dd);
  k_gemm<<<2,   256, 0, stream>>>(know16, Wq4t, bq4, nullptr, qkn16, NKd);
  k_gemm<<<128, 256, 0, stream>>>(hq16, Wk4t, bk4, nullptr, k416, M13);
  k_gemm<<<128, 256, 0, stream>>>(p16,  Wv4t, bv4, nullptr, v416, M13);
  // 64 batches x 8 heads; Q row = know query n=bt%16; K/V batch = bt/16
  k_attn<<<BSd * NKd * Hd, 256, SMEM, stream>>>(qkn16, k416, v416, gm4, big, ks_out,
                                                BSd * NKd, NKd, 1, 0, 1, 2);
  k_cast<<<(int)((ACT4 + 255) / 256), 256, 0, stream>>>(big, attn416, (int)ACT4);
  // output projection reuses `big` (attn4F consumed by the cast above)
  k_gemm<<<2048, 256, 0, stream>>>(attn416, Wo4t, bo4, big, nullptr, M4);
  // residual with broadcast know_params + LN + scatter into z [4,256,4096]
  k_ln<<<2048, 256, 0, stream>>>(know, big, lg4, lb4, z_out, nullptr, M4, 1);

  (void)ws_size;
}
